// CellGraph_4011499455036
// MI455X (gfx1250) — matrix operand order repaired and checked by reading, not compiled, not run on hardware
//
#include <hip/hip_runtime.h>
#include <hip/hip_bf16.h>
#include <math.h>

#define NN   10000
#define FIN  2000
#define HD   128
#define LDIM 15
#define AD   32
#define GD   2000
#define ED   320000

#define KT1  63          // ceil(FIN/32), zero-padded
#define KT2  32          // 1024/32
#define CT1  8           // HD/16
#define CTB  125         // GD/16
#define MT   (NN / 16)   // 625 row tiles

typedef __bf16 bf16_t;
typedef __attribute__((ext_vector_type(16))) __bf16 v16bf;
typedef __attribute__((ext_vector_type(8)))  float  v8f;

// ---------- helpers ----------

__device__ __forceinline__ bf16_t f2bf(float f) {
  unsigned u = __float_as_uint(f);
  unsigned r = (u + 0x7FFFu + ((u >> 16) & 1u)) >> 16;   // RNE
  unsigned short h = (unsigned short)r;
  bf16_t b;
  __builtin_memcpy(&b, &h, sizeof(b));
  return b;
}

// order-preserving float <-> uint map for atomicMax on floats
__device__ __forceinline__ unsigned fmap(float f) {
  unsigned b = __float_as_uint(f);
  return (b & 0x80000000u) ? ~b : (b | 0x80000000u);
}
__device__ __forceinline__ float funmap(unsigned u) {
  return (u & 0x80000000u) ? __uint_as_float(u & 0x7FFFFFFFu) : __uint_as_float(~u);
}

// A-fragment slot for element (row, kk) of a 16x32 bf16 tile.
// ISA 16-bit A layout: m = lane&15, k = (i>>3)*16 + (i&7) + 8*(lane>>4)
__device__ __forceinline__ int a_slot(int row, int kk) {
  int lane = row + (((kk >> 3) & 1) << 4);
  int i = ((kk >> 4) << 3) | (kk & 7);
  return lane * 16 + i;
}

__device__ __forceinline__ v8f wmma_bf16(v16bf a, v16bf b, v8f c) {
  return __builtin_amdgcn_wmma_f32_16x16x32_bf16(false, a, false, b, (short)0, c, false, false);
}

// ---------- pack kernels (one pass each, cheap) ----------

// Pack W[outdim, K] (row-major f32) into bf16 B-fragment tiles:
// tile (ct, kt) = 512 contiguous bf16; lane: col = lane&15, k = kt*32 + 16*(i>>3) + (i&7) + 8*(lane>>4)
__global__ __launch_bounds__(256) void k_packW(const float* __restrict__ W,
                                               bf16_t* __restrict__ out,
                                               int ctN, int K, int numKt) {
  int t = blockIdx.x * blockDim.x + threadIdx.x;
  int total = ctN * numKt * 512;
  if (t >= total) return;
  int tile = t >> 9, r = t & 511;
  int lane = r >> 4, i = r & 15;
  int ct = tile / numKt, kt = tile % numKt;
  int col = lane & 15;
  int k = kt * 32 + ((i >> 3) << 4) + (i & 7) + ((lane >> 4) << 3);
  int h = ct * 16 + col;
  out[t] = (k < K) ? f2bf(W[(size_t)h * K + k]) : f2bf(0.0f);
}

// Pack x[N, FIN] into bf16 A-fragment tiles (zero-padded past FIN)
__global__ __launch_bounds__(256) void k_packX(const float* __restrict__ x,
                                               bf16_t* __restrict__ out) {
  int t = blockIdx.x * blockDim.x + threadIdx.x;
  const int total = MT * KT1 * 512;
  if (t >= total) return;
  int tile = t >> 9, r = t & 511;
  int lane = r >> 4, i = r & 15;
  int mt = tile / KT1, kt = tile % KT1;
  int n = mt * 16 + (lane & 15);
  int k = kt * 32 + ((i >> 3) << 4) + (i & 7) + ((lane >> 4) << 3);
  out[t] = (k < FIN) ? f2bf(x[(size_t)n * FIN + k]) : f2bf(0.0f);
}

// ---------- simple kernels ----------

__global__ __launch_bounds__(256) void k_zero(float* __restrict__ p, long long n) {
  long long i = (long long)blockIdx.x * blockDim.x + threadIdx.x;
  if (i < n) p[i] = 0.0f;
}

// ---------- layer-1 fused linears (bf16 WMMA, f32 accum, no LDS) ----------
// Per K-tile: 1 A-fragment load + 4 B-fragment loads + 4 WMMAs.

__global__ __launch_bounds__(256) void k_lin1_wmma(
    const bf16_t* __restrict__ Xp,
    const bf16_t* __restrict__ wqp, const bf16_t* __restrict__ wkp,
    const bf16_t* __restrict__ wvp, const bf16_t* __restrict__ wsp,
    const float* __restrict__ bq, const float* __restrict__ bk,
    const float* __restrict__ bv, const float* __restrict__ bsk,
    float* __restrict__ q, float* __restrict__ k,
    float* __restrict__ v, float* __restrict__ s) {
  const int tid  = threadIdx.x;
  const int lane = tid & 31;
  const int wave = tid >> 5;           // = ct (HD = 8 waves * 16 cols)
  const int mt   = blockIdx.x;
  v8f aq = {}, ak = {}, av = {}, as = {};
  const v16bf* Af = (const v16bf*)(Xp + (size_t)mt * KT1 * 512);
  const v16bf* Bq = (const v16bf*)(wqp + (size_t)wave * KT1 * 512);
  const v16bf* Bk = (const v16bf*)(wkp + (size_t)wave * KT1 * 512);
  const v16bf* Bv = (const v16bf*)(wvp + (size_t)wave * KT1 * 512);
  const v16bf* Bs = (const v16bf*)(wsp + (size_t)wave * KT1 * 512);
  for (int kt = 0; kt < KT1; ++kt) {
    int fi = kt * 32 + lane;           // fragment slot index (32 v16bf per tile)
    v16bf a  = Af[fi];
    v16bf b0 = Bq[fi];
    v16bf b1 = Bk[fi];
    v16bf b2 = Bv[fi];
    v16bf b3 = Bs[fi];
    if (kt + 1 < KT1) __builtin_prefetch(&Af[fi + 32], 0, 3);
    aq = wmma_bf16(a, b0, aq);
    ak = wmma_bf16(a, b1, ak);
    av = wmma_bf16(a, b2, av);
    as = wmma_bf16(a, b3, as);
  }
  int col   = lane & 15;
  int rbase = (lane >> 4) * 8;
  int hc    = wave * 16 + col;
  const int m0 = mt * 16;
#pragma unroll
  for (int r = 0; r < 8; ++r) {
    size_t o = (size_t)(m0 + rbase + r) * HD + hc;
    q[o] = aq[r] + bq[hc];
    k[o] = ak[r] + bk[hc];
    v[o] = av[r] + bv[hc];
    s[o] = as[r] + bsk[hc];
  }
}

// ---------- attention (templated D: fully unrolled dots) ----------

template <int D>
__global__ __launch_bounds__(256) void k_att_logit(
    const float* __restrict__ q, const float* __restrict__ k,
    const int* __restrict__ src, const int* __restrict__ dst,
    float* __restrict__ logit, unsigned* __restrict__ m_u, float scale) {
  int e = blockIdx.x * blockDim.x + threadIdx.x;
  if (e >= ED) return;
  int se = src[e], te = dst[e];
  const float* qr = q + (size_t)te * D;
  const float* kr = k + (size_t)se * D;
  float acc = 0.0f;
#pragma unroll
  for (int i = 0; i < D; ++i) acc += qr[i] * kr[i];
  acc *= scale;
  logit[e] = acc;
  atomicMax(m_u + te, fmap(acc));
}

__global__ __launch_bounds__(256) void k_att_exp(
    const float* __restrict__ logit, const unsigned* __restrict__ m_u,
    const int* __restrict__ dst, float* __restrict__ ex, float* __restrict__ den) {
  int e = blockIdx.x * blockDim.x + threadIdx.x;
  if (e >= ED) return;
  int te = dst[e];
  float x = __expf(logit[e] - funmap(m_u[te]));
  ex[e] = x;
  atomicAdd(den + te, x);
}

template <int D>
__global__ __launch_bounds__(256) void k_att_agg(
    const float* __restrict__ ex, const float* __restrict__ den,
    const float* __restrict__ v, const int* __restrict__ src, const int* __restrict__ dst,
    float* __restrict__ agg) {
  long long t = (long long)blockIdx.x * blockDim.x + threadIdx.x;
  if (t >= (long long)ED * D) return;
  int e = (int)(t / D);
  int f = (int)(t % D);
  int se = src[e], te = dst[e];
  float attn = ex[e] / den[te];
  atomicAdd(agg + (size_t)te * D + f, attn * v[(size_t)se * D + f]);
}

__global__ __launch_bounds__(256) void k_addskip(
    const float* __restrict__ agg, const float* __restrict__ skip,
    float* __restrict__ out, long long n, int dorelu) {
  long long i = (long long)blockIdx.x * blockDim.x + threadIdx.x;
  if (i >= n) return;
  float x = agg[i] + skip[i];
  out[i] = dorelu ? fmaxf(x, 0.0f) : x;
}

// ---------- layer-2 fused linears (tiny N/K -> VALU) ----------

__global__ __launch_bounds__(256) void k_lin2(
    const float* __restrict__ h,
    const float* __restrict__ wq, const float* __restrict__ bq,
    const float* __restrict__ wk, const float* __restrict__ bk,
    const float* __restrict__ wv, const float* __restrict__ bv,
    const float* __restrict__ wsk, const float* __restrict__ bsk,
    float* __restrict__ q, float* __restrict__ k,
    float* __restrict__ v, float* __restrict__ s) {
  int t = blockIdx.x * blockDim.x + threadIdx.x;
  if (t >= NN * LDIM) return;
  int n = t / LDIM, l = t % LDIM;
  const float* hr = h + (size_t)n * HD;
  float aq = bq[l], ak = bk[l], av = bv[l], as = bsk[l];
#pragma unroll 8
  for (int i = 0; i < HD; ++i) {
    float xv = hr[i];
    aq += xv * wq[l * HD + i];
    ak += xv * wk[l * HD + i];
    av += xv * wv[l * HD + i];
    as += xv * wsk[l * HD + i];
  }
  q[t] = aq; k[t] = ak; v[t] = av; s[t] = as;
}

// hh = z @ wlin.T + blin   (N x AD)
__global__ __launch_bounds__(256) void k_hh(
    const float* __restrict__ z, const float* __restrict__ wlin,
    const float* __restrict__ blin, float* __restrict__ hh) {
  int t = blockIdx.x * blockDim.x + threadIdx.x;
  if (t >= NN * AD) return;
  int n = t >> 5, a = t & 31;
  const float* zr = z + (size_t)n * LDIM;
  float acc = blin[a];
#pragma unroll
  for (int l = 0; l < LDIM; ++l) acc += zr[l] * wlin[a * LDIM + l];
  hh[t] = acc;
}

// P[n, i*32+j] = hh[n,i]*hh[n,j], written directly in packed A-fragment layout
__global__ __launch_bounds__(256) void k_outer(
    const float* __restrict__ hh, bf16_t* __restrict__ Pp) {
  int t = blockIdx.x * blockDim.x + threadIdx.x;
  if (t >= NN * 1024) return;
  int n = t >> 10, p = t & 1023;
  int i = p >> 5, j = p & 31;              // k-dim index p = i*32+j
  int mt = n >> 4, row = n & 15;
  Pp[((size_t)mt * KT2 + i) * 512 + a_slot(row, j)] =
      f2bf(hh[n * AD + i] * hh[n * AD + j]);
}

// ---------- bilinear decoder GEMM: dec = sigmoid(P @ wbil.T + b) ----------
// A and B both pre-packed; each wave: 1 M-tile x 4 N-tiles -> 4 WMMAs per A load.

__global__ __launch_bounds__(256) void k_bilinear_wmma(
    const bf16_t* __restrict__ Pp, const bf16_t* __restrict__ Wbp,
    const float* __restrict__ bbil, float* __restrict__ dec) {
  const int tid  = threadIdx.x;
  const int lane = tid & 31;
  const int wave = tid >> 5;
  const int mt = blockIdx.x;
  const int tbase = blockIdx.y * 32 + wave * 4;   // first of 4 col tiles (of 125)
  const bool c1 = (tbase + 1 < CTB), c2 = (tbase + 2 < CTB), c3 = (tbase + 3 < CTB);
  if (tbase >= CTB) return;
  v8f acc0 = {}, acc1 = {}, acc2 = {}, acc3 = {};
  const v16bf* Af = (const v16bf*)(Pp + (size_t)mt * KT2 * 512);
  const v16bf* B0 = (const v16bf*)(Wbp + (size_t)(tbase + 0) * KT2 * 512);
  const v16bf* B1 = (const v16bf*)(Wbp + (size_t)(tbase + 1) * KT2 * 512);
  const v16bf* B2 = (const v16bf*)(Wbp + (size_t)(tbase + 2) * KT2 * 512);
  const v16bf* B3 = (const v16bf*)(Wbp + (size_t)(tbase + 3) * KT2 * 512);
  for (int kt = 0; kt < KT2; ++kt) {
    int fi = kt * 32 + lane;
    v16bf a = Af[fi];
    if (kt + 1 < KT2) __builtin_prefetch(&Af[fi + 32], 0, 3);
    {
      v16bf b = B0[fi];
      acc0 = wmma_bf16(a, b, acc0);
    }
    if (c1) {
      v16bf b = B1[fi];
      acc1 = wmma_bf16(a, b, acc1);
    }
    if (c2) {
      v16bf b = B2[fi];
      acc2 = wmma_bf16(a, b, acc2);
    }
    if (c3) {
      v16bf b = B3[fi];
      acc3 = wmma_bf16(a, b, acc3);
    }
  }
  const int col   = lane & 15;
  const int rbase = (lane >> 4) * 8;
  const int m0 = mt * 16;
  v8f* accs[4] = {&acc0, &acc1, &acc2, &acc3};
#pragma unroll
  for (int j = 0; j < 4; ++j) {
    int ct = tbase + j;
    if (ct >= CTB) break;
    int g = ct * 16 + col;
    float bb = bbil[g];
    v8f acc = *accs[j];
#pragma unroll
    for (int r = 0; r < 8; ++r) {
      float val = acc[r] + bb;
      dec[(size_t)(m0 + rbase + r) * GD + g] = 1.0f / (1.0f + __expf(-val));
    }
  }
}

// ---------- launch ----------

extern "C" void kernel_launch(void* const* d_in, const int* in_sizes, int n_in,
                              void* d_out, int out_size, void* d_ws, size_t ws_size,
                              hipStream_t stream) {
  const float* x    = (const float*)d_in[0];
  const int*   eidx = (const int*)d_in[1];
  const float* wq1 = (const float*)d_in[2];  const float* bq1 = (const float*)d_in[3];
  const float* wk1 = (const float*)d_in[4];  const float* bk1 = (const float*)d_in[5];
  const float* wv1 = (const float*)d_in[6];  const float* bv1 = (const float*)d_in[7];
  const float* ws1 = (const float*)d_in[8];  const float* bs1 = (const float*)d_in[9];
  const float* wq2 = (const float*)d_in[10]; const float* bq2 = (const float*)d_in[11];
  const float* wk2 = (const float*)d_in[12]; const float* bk2 = (const float*)d_in[13];
  const float* wv2 = (const float*)d_in[14]; const float* bv2 = (const float*)d_in[15];
  const float* ws2 = (const float*)d_in[16]; const float* bs2 = (const float*)d_in[17];
  const float* wlin = (const float*)d_in[18]; const float* blin = (const float*)d_in[19];
  const float* wbil = (const float*)d_in[20]; const float* bbil = (const float*)d_in[21];
  const int* src = eidx;
  const int* dst = eidx + ED;

  float* dec  = (float*)d_out;                 // [N, G]
  float* zout = dec + (size_t)NN * GD;         // [N, L] (z_mean)

  char* ws = (char*)d_ws;
  size_t off = 0;
  auto alloc = [&](size_t bytes) -> void* {
    void* p = ws + off;
    off += (bytes + 255) & ~(size_t)255;
    return p;
  };
  float* q1   = (float*)alloc((size_t)NN * HD * 4);
  float* k1   = (float*)alloc((size_t)NN * HD * 4);
  float* v1   = (float*)alloc((size_t)NN * HD * 4);
  float* s1   = (float*)alloc((size_t)NN * HD * 4);
  float* agg1 = (float*)alloc((size_t)NN * HD * 4);
  float* h    = (float*)alloc((size_t)NN * HD * 4);
  float* logit = (float*)alloc((size_t)ED * 4);
  float* exb   = (float*)alloc((size_t)ED * 4);
  unsigned* m_u = (unsigned*)alloc((size_t)NN * 4);
  float* den   = (float*)alloc((size_t)NN * 4);
  float* q2   = (float*)alloc((size_t)NN * LDIM * 4);
  float* k2   = (float*)alloc((size_t)NN * LDIM * 4);
  float* v2   = (float*)alloc((size_t)NN * LDIM * 4);
  float* s2   = (float*)alloc((size_t)NN * LDIM * 4);
  float* agg2 = (float*)alloc((size_t)NN * LDIM * 4);
  float* hh   = (float*)alloc((size_t)NN * AD * 4);
  const size_t wp1sz = (size_t)CT1 * KT1 * 512;      // per conv1 weight (elems)
  bf16_t* wq1p = (bf16_t*)alloc(wp1sz * 2);
  bf16_t* wk1p = (bf16_t*)alloc(wp1sz * 2);
  bf16_t* wv1p = (bf16_t*)alloc(wp1sz * 2);
  bf16_t* ws1p = (bf16_t*)alloc(wp1sz * 2);

  // Xp is only live during phase 1 (packX + lin1); Pp/Wbp alias into the same region.
  const size_t xpsz  = (size_t)MT * KT1 * 512;       // elems
  const size_t ppsz  = (size_t)MT * KT2 * 512;       // elems
  const size_t wbpsz = (size_t)CTB * KT2 * 512;      // elems
  char* region = (char*)alloc(xpsz * 2);             // 40.3 MB >= ppsz*2 + wbpsz*2
  bf16_t* Xp  = (bf16_t*)region;
  bf16_t* Pp  = (bf16_t*)region;
  bf16_t* wbp = (bf16_t*)(region + ((ppsz * 2 + 255) & ~(size_t)255));

  auto cdiv = [](long long a, long long b) { return (int)((a + b - 1) / b); };

  // ----- pack conv1 weights + x to bf16 fragment layout -----
  k_packW<<<cdiv((long long)wp1sz, 256), 256, 0, stream>>>(wq1, wq1p, CT1, FIN, KT1);
  k_packW<<<cdiv((long long)wp1sz, 256), 256, 0, stream>>>(wk1, wk1p, CT1, FIN, KT1);
  k_packW<<<cdiv((long long)wp1sz, 256), 256, 0, stream>>>(wv1, wv1p, CT1, FIN, KT1);
  k_packW<<<cdiv((long long)wp1sz, 256), 256, 0, stream>>>(ws1, ws1p, CT1, FIN, KT1);
  k_packX<<<cdiv((long long)xpsz, 256), 256, 0, stream>>>(x, Xp);

  // ----- layer 1: TransformerConv(FIN -> HD) -----
  k_zero<<<cdiv((long long)NN * HD, 256), 256, 0, stream>>>(agg1, (long long)NN * HD);
  k_zero<<<cdiv(NN, 256), 256, 0, stream>>>((float*)m_u, NN);
  k_zero<<<cdiv(NN, 256), 256, 0, stream>>>(den, NN);

  k_lin1_wmma<<<MT, 256, 0, stream>>>(Xp, wq1p, wk1p, wv1p, ws1p,
                                      bq1, bk1, bv1, bs1, q1, k1, v1, s1);

  k_att_logit<HD><<<cdiv(ED, 256), 256, 0, stream>>>(q1, k1, src, dst, logit, m_u,
                                                     rsqrtf((float)HD));
  k_att_exp<<<cdiv(ED, 256), 256, 0, stream>>>(logit, m_u, dst, exb, den);
  k_att_agg<HD><<<cdiv((long long)ED * HD, 256), 256, 0, stream>>>(exb, den, v1, src, dst, agg1);
  k_addskip<<<cdiv((long long)NN * HD, 256), 256, 0, stream>>>(agg1, s1, h,
                                                               (long long)NN * HD, 1);

  // ----- layer 2: TransformerConv(HD -> LDIM) -----
  k_zero<<<cdiv((long long)NN * LDIM, 256), 256, 0, stream>>>(agg2, (long long)NN * LDIM);
  k_zero<<<cdiv(NN, 256), 256, 0, stream>>>((float*)m_u, NN);
  k_zero<<<cdiv(NN, 256), 256, 0, stream>>>(den, NN);

  k_lin2<<<cdiv((long long)NN * LDIM, 256), 256, 0, stream>>>(h, wq2, bq2, wk2, bk2,
                                                              wv2, bv2, ws2, bs2,
                                                              q2, k2, v2, s2);
  k_att_logit<LDIM><<<cdiv(ED, 256), 256, 0, stream>>>(q2, k2, src, dst, logit, m_u,
                                                       rsqrtf((float)LDIM));
  k_att_exp<<<cdiv(ED, 256), 256, 0, stream>>>(logit, m_u, dst, exb, den);
  k_att_agg<LDIM><<<cdiv((long long)ED * LDIM, 256), 256, 0, stream>>>(exb, den, v2, src, dst, agg2);
  k_addskip<<<cdiv((long long)NN * LDIM, 256), 256, 0, stream>>>(agg2, s2, zout,
                                                                 (long long)NN * LDIM, 0);

  // ----- decoder (Xp dead; Pp/wbp reuse its region) -----
  k_hh<<<cdiv((long long)NN * AD, 256), 256, 0, stream>>>(zout, wlin, blin, hh);
  k_packW<<<cdiv((long long)wbpsz, 256), 256, 0, stream>>>(wbil, wbp, CTB, 1024, KT2);
  k_outer<<<cdiv((long long)NN * 1024, 256), 256, 0, stream>>>(hh, Pp);
  dim3 bgrid(MT, 4);
  k_bilinear_wmma<<<bgrid, 256, 0, stream>>>(Pp, wbp, bbil, dec);
}
